// FT_QuantLinear_89885075570939
// MI455X (gfx1250) — matrix operand order repaired and checked by reading, not compiled, not run on hardware
//
#include <hip/hip_runtime.h>
#include <hip/hip_bf16.h>
#include <cstdint>

typedef __attribute__((ext_vector_type(16))) _Float16 v16h;
typedef __attribute__((ext_vector_type(8)))  _Float16 v8h;
typedef __attribute__((ext_vector_type(4)))  _Float16 v4h;
typedef __attribute__((ext_vector_type(2)))  _Float16 v2h;
typedef __attribute__((ext_vector_type(8)))  float    v8f;
typedef __attribute__((ext_vector_type(4)))  unsigned v4u;
typedef __attribute__((ext_vector_type(8)))  int      v8i;
typedef __attribute__((ext_vector_type(4)))  int      v4i;

#define KDIM   4096
#define NDIM   11008
#define BM     128
#define BN     128
#define BK     32
#define KTILES (KDIM / BK)
#define LDA    40            // padded LDS row stride (f16): 80 B
#define LDW    40
#define ABUF_BYTES (BM * LDA * 2)            // 10240
#define WBUF_BYTES (BN * LDW * 2)            // 10240
#define SMEM_BYTES (2 * ABUF_BYTES + 2 * WBUF_BYTES)   // 40960

// ---- TDM: async-load one 128x32 f16 tile of x into LDS, rows padded to 40 halves ----
__device__ __forceinline__ void tdm_load_x_tile(const _Float16* x, int m0, int rows_left,
                                                int kt, unsigned lds_off) {
    unsigned long long ga =
        (unsigned long long)(uintptr_t)(x + (size_t)m0 * KDIM + (size_t)kt * BK);
    // D# group 0: count=1 | lds_addr | global_addr(57b) | type=2
    v4u g0 = { 1u,
               lds_off,
               (unsigned)(ga & 0xFFFFFFFFu),
               (unsigned)((ga >> 32) & 0x1FFFFFFu) | (2u << 30) };
    // D# group 1:
    //  dw0: data_size=1(2B)<<16 | pad_enable<<20 | pad_interval=3(16 dw)<<22 | pad_amount=3(4 dw)<<25
    //  dw1[31:16]=tensor_dim0.lo16 (=K)        dw2[31:16]=tensor_dim1.lo16 (rows)
    //  dw3[31:16]=tile_dim0 (=BK)              dw4[15:0]=tile_dim1 (=BM)
    //  dw5=tensor_dim0_stride.lo32 (=K)
    v8i g1 = { (int)((1u << 16) | (1u << 20) | (3u << 22) | (3u << 25)),
               (int)((unsigned)KDIM << 16),
               (int)((unsigned)rows_left << 16),
               (int)((unsigned)BK << 16),
               (int)BM,
               (int)KDIM,
               0, 0 };
    v4i z4 = { 0, 0, 0, 0 };
    v8i z8 = { 0, 0, 0, 0, 0, 0, 0, 0 };
    __builtin_amdgcn_tensor_load_to_lds(g0, g1, z4, z4, z8, 0);
}

__global__ __launch_bounds__(256) void FT_QuantLinear_89885075570939_kernel(
    const _Float16* __restrict__ x,        // (M, K) f16
    const int*      __restrict__ qweight,  // (N/4, K) packed uint16-in-int32
    const _Float16* __restrict__ scales,   // (NG, N) f16
    const _Float16* __restrict__ zeros,    // (NG, N) f16 (stored -zero*scale)
    const _Float16* __restrict__ bias,     // (N,) f16
    float*          __restrict__ out,      // (M, N) f32
    int M)
{
    extern __shared__ char smem[];         // dynamic LDS => base offset 0 for TDM D#

    const int tid  = threadIdx.x;
    const int lane = tid & 31;
    const int wid  = tid >> 5;     // 0..7
    const int wm   = wid >> 2;     // 0..1 : 64-row slab
    const int wn   = wid & 3;      // 0..3 : 32-col slab

    const int m0 = blockIdx.y * BM;
    const int n0 = blockIdx.x * BN;

    v8f acc[4][2] = {};

    // ---- per-thread dequant mapping (constant across K loop) ----
    const int nl    = tid >> 1;          // local n: 0..127 (2 threads per n)
    const int n     = n0 + nl;
    const int a4    = n & 3;             // INTER nibble-group index
    const int wsel  = tid & 1;           // which 4-of-8 packed words this thread decodes
    const int* qrow = qweight + (size_t)(n >> 2) * KDIM;

    // packed-word column base for K-tile kt (16B-aligned: multiple of 4 words)
    auto qcol = [&](int kt) -> int {
        const int k0 = kt * BK;
        return (k0 & ~63) + a4 * 16 + ((k0 & 63) >> 2) + wsel * 4;
    };

    // ---- prologue: TDM tile 0 + first qweight words ----
    if (wid == 0) tdm_load_x_tile(x, m0, M - m0, 0, 0u);
    v4i qw = *(const v4i*)(qrow + qcol(0));

    _Float16 scv = (_Float16)0.0f, zpv = (_Float16)0.0f;

    for (int kt = 0; kt < KTILES; ++kt) {
        const int cur = kt & 1;
        _Float16* Ac = (_Float16*)(smem + cur * ABUF_BYTES);
        _Float16* Wc = (_Float16*)(smem + 2 * ABUF_BYTES + cur * WBUF_BYTES);

        // ---- pipeline next tile: TDM A-load + qweight regs ----
        v4i qw_next = qw;
        if (kt + 1 < KTILES) {
            if (wid == 0)
                tdm_load_x_tile(x, m0, M - m0, kt + 1, (unsigned)((cur ^ 1) * ABUF_BYTES));
            qw_next = *(const v4i*)(qrow + qcol(kt + 1));
        }

        // ---- per-group scale/zero (group = 128 k = 4 tiles); fold 1024*sc into zp ----
        if ((kt & 3) == 0) {
            const int g  = kt >> 2;
            const float sc = (float)scales[(size_t)g * NDIM + n];
            const float zp = (float)zeros [(size_t)g * NDIM + n];
            scv = (_Float16)sc;
            zpv = (_Float16)(zp - 1024.0f * sc);
        }
        const v2h sc2 = { scv, scv };
        const v2h zp2 = { zpv, zpv };

        // ---- dequant: nibble i of word w -> Wc[nl][i*8 + wsel*4 + w] ----
        #pragma unroll
        for (int i = 0; i < 4; ++i) {
            const int sh = 4 * i;
            unsigned u0 = 0x6400u | (((unsigned)qw.x >> sh) & 0xFu);
            unsigned u1 = 0x6400u | (((unsigned)qw.y >> sh) & 0xFu);
            unsigned u2 = 0x6400u | (((unsigned)qw.z >> sh) & 0xFu);
            unsigned u3 = 0x6400u | (((unsigned)qw.w >> sh) & 0xFu);
            v2h h01 = __builtin_bit_cast(v2h, (unsigned)(u0 | (u1 << 16)));
            v2h h23 = __builtin_bit_cast(v2h, (unsigned)(u2 | (u3 << 16)));
            v2h r01 = h01 * sc2 + zp2;     // v_pk_fma_f16
            v2h r23 = h23 * sc2 + zp2;
            v4h res;
            res[0] = r01[0]; res[1] = r01[1]; res[2] = r23[0]; res[3] = r23[1];
            *(v4h*)(&Wc[nl * LDW + i * 8 + wsel * 4]) = res;   // ds_store_b64
        }

        // ---- wait for current A tile (TDM completes in order), then sync ----
        if (wid == 0) {
            if (kt + 1 < KTILES) __builtin_amdgcn_s_wait_tensorcnt((short)1);
            else                 __builtin_amdgcn_s_wait_tensorcnt((short)0);
        }
        __syncthreads();

        // ---- 4x2 WMMA per wave ----
        #pragma unroll
        for (int tm = 0; tm < 4; ++tm) {
            const int arow  = wm * 64 + tm * 16 + (lane & 15);
            const int kbase = (lane < 16) ? 0 : 8;
            v8h alo = *(const v8h*)(&Ac[arow * LDA + kbase]);
            v8h ahi = *(const v8h*)(&Ac[arow * LDA + kbase + 16]);
            v16h afrag;
            #pragma unroll
            for (int i = 0; i < 8; ++i) { afrag[i] = alo[i]; afrag[8 + i] = ahi[i]; }

            #pragma unroll
            for (int tn = 0; tn < 2; ++tn) {
                const int bcol = wn * 32 + tn * 16 + (lane & 15);
                const int kb2  = (lane < 16) ? 0 : 8;
                v8h blo = *(const v8h*)(&Wc[bcol * LDW + kb2]);
                v8h bhi = *(const v8h*)(&Wc[bcol * LDW + kb2 + 16]);
                v16h bfrag;
                #pragma unroll
                for (int i = 0; i < 8; ++i) { bfrag[i] = blo[i]; bfrag[8 + i] = bhi[i]; }

                acc[tm][tn] = __builtin_amdgcn_wmma_f32_16x16x32_f16(
                    false, afrag, false, bfrag, (short)0, acc[tm][tn], false, false);
            }
        }

        __syncthreads();
        qw = qw_next;
    }

    // ---- epilogue: C/D layout -> global f32 stores, add bias ----
    #pragma unroll
    for (int tm = 0; tm < 4; ++tm) {
        #pragma unroll
        for (int tn = 0; tn < 2; ++tn) {
            const int ncol = n0 + wn * 32 + tn * 16 + (lane & 15);
            const float bv = (float)bias[ncol];
            const int mrow0 = m0 + wm * 64 + tm * 16 + ((lane >> 4) << 3);
            #pragma unroll
            for (int r = 0; r < 8; ++r) {
                out[(size_t)(mrow0 + r) * NDIM + ncol] = acc[tm][tn][r] + bv;
            }
        }
    }
}

extern "C" void kernel_launch(void* const* d_in, const int* in_sizes, int n_in,
                              void* d_out, int out_size, void* d_ws, size_t ws_size,
                              hipStream_t stream) {
    (void)n_in; (void)d_ws; (void)ws_size; (void)out_size;

    const _Float16* x       = (const _Float16*)d_in[0];
    const int*      qweight = (const int*)     d_in[1];
    const _Float16* scales  = (const _Float16*)d_in[2];
    const _Float16* zeros   = (const _Float16*)d_in[3];
    const _Float16* bias    = (const _Float16*)d_in[4];
    float*          out     = (float*)d_out;

    const int M = in_sizes[0] / KDIM;   // B*S = 8192

    dim3 grid(NDIM / BN, M / BM);       // (86, 64)
    dim3 block(256);
    FT_QuantLinear_89885075570939_kernel<<<grid, block, SMEM_BYTES, stream>>>(
        x, qweight, scales, zeros, bias, out, M);
}
